// DeformableBottleneck_35897336660548
// MI455X (gfx1250) — compile-verified, matrix operand order checked by reading, not run on hardware
//
#include <hip/hip_runtime.h>

typedef __bf16 bf16;
typedef __attribute__((ext_vector_type(16))) __bf16 v16bf;
typedef __attribute__((ext_vector_type(8)))  __bf16 v8bf;
typedef __attribute__((ext_vector_type(8)))  float  v8f;
typedef __attribute__((ext_vector_type(4)))  unsigned v4u;
typedef __attribute__((ext_vector_type(8)))  int    v8i;
typedef __attribute__((ext_vector_type(4)))  int    v4i;

#define CIN  1024
#define P_   256
#define O_   1024
#define HW   1024      // 32*32
#define M_   16384     // B*H*W
#define K2   2304      // 9*P_
#define MT   128       // block tile rows (all GEMMs)
#define SP   136       // padded stage stride (bf16 elems)

#if __has_builtin(__builtin_amdgcn_tensor_load_to_lds)
#  if (__clang_major__ <= 22)
#    define USE_TDM 1          // ROCm 7.2: 5-arg form
#  else
#    define USE_TDM 2          // clang-23 lane: 6-arg form
#  endif
#else
#  define USE_TDM 0
#endif

#define WMMA(a,b,c) __builtin_amdgcn_wmma_f32_16x16x32_bf16(false,(a),false,(b),(short)0,(c),false,false)

// ---------- bf16 helpers ----------
__device__ inline unsigned f2bf_bits(float f) {
  unsigned u = __builtin_bit_cast(unsigned, f);
  return (u + 0x7FFFu + ((u >> 16) & 1u)) >> 16;   // RNE
}
__device__ inline bf16 f2bf(float f) {
  return __builtin_bit_cast(bf16, (unsigned short)f2bf_bits(f));
}
__device__ inline unsigned pack2(float lo, float hi) {
  return (f2bf_bits(lo) & 0xFFFFu) | (f2bf_bits(hi) << 16);
}
__device__ inline float bf_lo(unsigned u) { return __builtin_bit_cast(float, u << 16); }
__device__ inline float bf_hi(unsigned u) { return __builtin_bit_cast(float, u & 0xFFFF0000u); }

// ---------- CDNA5 async DMA: global <-> LDS, 16B per lane (ASYNCcnt) ----------
__device__ inline void async_ld16(bf16* lp, const bf16* gp) {
  unsigned            lo = (unsigned)(size_t)lp;
  unsigned long long  ga = (unsigned long long)(size_t)gp;
  asm volatile("global_load_async_to_lds_b128 %0, %1, off" :: "v"(lo), "v"(ga) : "memory");
}
__device__ inline void async_st16(const bf16* lp, bf16* gp) {
  unsigned            lo = (unsigned)(size_t)lp;
  unsigned long long  ga = (unsigned long long)(size_t)gp;
  asm volatile("global_store_async_from_lds_b128 %0, %1, off" :: "v"(ga), "v"(lo) : "memory");
}
template<int N>
__device__ inline void wait_async_n() {
  asm volatile("s_wait_asynccnt %0" :: "i"(N) : "memory");
}

// ---------- CDNA5 TDM: one descriptor-driven 2D tile DMA (TENSORcnt) ----------
// rows x cols (data_size=2B) tile, row stride = rowstride elems, dense into LDS.
__device__ inline void tdm_load_2d(bf16* lds, const bf16* gsrc, int rows, int cols, int rowstride) {
#if USE_TDM
  unsigned long long ga = (unsigned long long)(size_t)gsrc;
  v4u g0; v8i g1;
  v4i g2 = {0, 0, 0, 0}, g3 = {0, 0, 0, 0};
  g0[0] = 1u;                                                   // count=1, user mode
  g0[1] = (unsigned)(size_t)lds;                                // lds_addr (bytes)
  g0[2] = (unsigned)ga;                                         // global_addr[31:0]
  g0[3] = ((unsigned)(ga >> 32) & 0x01FFFFFFu) | 0x80000000u;   // addr[56:32] | type=2
  g1[0] = 1 << 16;            // workgroup_mask=0 | data_size=1 (2 bytes)
  g1[1] = 0;                  // no atomic barrier; tensor_dim0[15:0]=0 (dim0 = 1<<20)
  g1[2] = 0x10;               // tensor_dim0[31:16]=16 ; tensor_dim1[15:0]=0
  g1[3] = 0x10 | (cols << 16);// tensor_dim1[31:16]=16 ; tile_dim0=cols
  g1[4] = rows;               // tile_dim1=rows ; tile_dim2=0
  g1[5] = rowstride;          // tensor_dim0_stride[31:0]
  g1[6] = 0;                  // stride hi ; tensor_dim1_stride lo
  g1[7] = 0;
#  if USE_TDM == 1
  __builtin_amdgcn_tensor_load_to_lds(g0, g1, g2, g3, 0);
#  else
  v8i g4 = {0, 0, 0, 0, 0, 0, 0, 0};
  __builtin_amdgcn_tensor_load_to_lds(g0, g1, g2, g3, g4, 0);
#  endif
#endif
}

// ---------- WMMA fragment from LDS tile stored row-major [row][32] ----------
__device__ inline v16bf frag_ld(const bf16* tile, int lane) {
  int row = lane & 15;
  int hl  = lane >> 4;
  const bf16* p = tile + row * 32 + hl * 8;
  v8bf a = *(const v8bf*)(p);
  v8bf b = *(const v8bf*)(p + 16);
  return __builtin_shufflevector(a, b, 0,1,2,3,4,5,6,7,8,9,10,11,12,13,14,15);
}

// ---------- tile DMA: ROWS x 32 bf16, source row-major stride K ----------
template<int ROWS>
__device__ inline void issue_tile(bf16* lds, const bf16* src, int row0, int K, int k0, int tid) {
#pragma unroll
  for (int j = 0; j < ROWS / 64; ++j) {
    int idx = j * 256 + tid;
    int rr = idx >> 2, sg = idx & 3;
    async_ld16(&lds[rr * 32 + sg * 8], src + (size_t)(row0 + rr) * K + k0 + sg * 8);
  }
}

// ---------- wave MMA: 32 rows x NQ*16 cols; B frags loaded one at a time ----------
template<int NQ>
__device__ inline void mma_tile(const bf16* bufA, const bf16* bufB, int lane,
                                int mig, int nig, v8f (&acc)[2][NQ]) {
  v16bf a0 = frag_ld(bufA + (mig * 2 + 0) * 512, lane);
  v16bf a1 = frag_ld(bufA + (mig * 2 + 1) * 512, lane);
#pragma unroll
  for (int q = 0; q < NQ; ++q) {
    v16bf bq = frag_ld(bufB + (nig * NQ + q) * 512, lane);
    acc[0][q] = WMMA(a0, bq, acc[0][q]);
    acc[1][q] = WMMA(a1, bq, acc[1][q]);
  }
}

// ---------- double-buffered GEMM mainloop (A,B bf16 row-major [row][K]) ----------
template<int NQ>   // B tile rows = NQ*32 (64 or 128)
__device__ inline void gemm_db(const bf16* __restrict__ srcA, const bf16* __restrict__ srcB,
                               int m0, int n0, int K, int tid,
                               bf16* A0, bf16* A1, bf16* B0, bf16* B1,
                               v8f (&acc)[2][NQ]) {
  const int lane = tid & 31, wv = tid >> 5;
  const int mig = wv & 3, nig = wv >> 2;
  constexpr int NROWS = NQ * 32;
  constexpr int NAS = 2 + NROWS / 64;    // asyncs per thread per chunk
  issue_tile<MT>(A0, srcA, m0, K, 0, tid);
  issue_tile<NROWS>(B0, srcB, n0, K, 0, tid);
  const int nch = K / 32;
#pragma unroll 2
  for (int i = 0; i < nch; ++i) {
    bf16* cA = (i & 1) ? A1 : A0;  bf16* cB = (i & 1) ? B1 : B0;
    bf16* nA = (i & 1) ? A0 : A1;  bf16* nB = (i & 1) ? B0 : B1;
    __syncthreads();                           // all reads of nA/nB finished
    if (i + 1 < nch) {
      issue_tile<MT>(nA, srcA, m0, K, (i + 1) * 32, tid);
      issue_tile<NROWS>(nB, srcB, n0, K, (i + 1) * 32, tid);
      wait_async_n<NAS>();                     // chunk i done; chunk i+1 in flight
    } else {
      wait_async_n<0>();
    }
    __syncthreads();                           // everyone's chunk-i DMA landed
    mma_tile<NQ>(cA, cB, lane, mig, nig, acc); // overlaps chunk i+1 DMA
  }
}

// ---------- staged bf16 epilogue: BN+ReLU -> LDS -> async b128 stores ----------
template<int NQ>
__device__ inline void epilogue_bnrelu(v8f (&acc)[2][NQ], bf16* stage,
                                       const float* sv, const float* tv,
                                       bf16* dst, int m0, int n0, int tid) {
  const int lane = tid & 31, wv = tid >> 5;
  const int mig = wv & 3, nig = wv >> 2;
  int hl = lane >> 4, col = lane & 15;
#pragma unroll
  for (int q = 0; q < NQ; ++q) {
    int cl = (nig * NQ + q) * 16 + col;
    float s = sv[n0 + cl], t = tv[n0 + cl];
#pragma unroll
    for (int p = 0; p < 2; ++p)
#pragma unroll
      for (int r = 0; r < 8; ++r) {
        int rl = (mig * 2 + p) * 16 + r + 8 * hl;
        stage[rl * SP + cl] = f2bf(fmaxf(acc[p][q][r] * s + t, 0.f));
      }
  }
  __syncthreads();
#pragma unroll
  for (int j = 0; j < 8; ++j) {                // 128x128 bf16, 16B per lane per step
    int idx = j * 256 + tid;
    int ml = idx >> 4, cs = (idx & 15) * 8;
    async_st16(&stage[ml * SP + cs], dst + (size_t)(m0 + ml) * P_ + n0 + cs);
  }
  wait_async_n<0>();
}

// ============================ prep kernels ============================
__global__ void k_bnprep(const float* g1,const float* bb1,const float* mm1,const float* vv1,
                         const float* g2,const float* bb2,const float* mm2,const float* vv2,
                         const float* g3,const float* bb3,const float* mm3,const float* vv3,
                         float* s1,float* t1,float* s2,float* t2,float* s3,float* t3) {
  int i = blockIdx.x * 256 + threadIdx.x;
  if (i < 256)       { float s = g1[i] * rsqrtf(vv1[i] + 1e-5f); s1[i] = s; t1[i] = bb1[i] - mm1[i] * s; }
  else if (i < 512)  { int j = i - 256; float s = g2[j] * rsqrtf(vv2[j] + 1e-5f); s2[j] = s; t2[j] = bb2[j] - mm2[j] * s; }
  else if (i < 1536) { int j = i - 512; float s = g3[j] * rsqrtf(vv3[j] + 1e-5f); s3[j] = s; t3[j] = bb3[j] - mm3[j] * s; }
}

__global__ void k_castw(const float* __restrict__ src, bf16* __restrict__ dst, int n) {
  int i = blockIdx.x * 256 + threadIdx.x;
  if (i < n) dst[i] = f2bf(src[i]);
}

// pack w2 (O=256,C=256,3,3) -> B2t[n][kidx*256+c] bf16 (tap-major K)
__global__ void k_packw2(const float* __restrict__ w2, bf16* __restrict__ B2t) {
  int i = blockIdx.x * 256 + threadIdx.x;
  if (i >= 256 * K2) return;
  int n = i / K2, r = i % K2;
  int kidx = r >> 8, c = r & 255;
  B2t[i] = f2bf(w2[(size_t)(n * 256 + c) * 9 + kidx]);
}

// transpose+convert x NCHW f32 -> xt[m][c] bf16 (channels-last), LDS-tiled 32x32
__global__ __launch_bounds__(256) void k_xt(const float* __restrict__ x, bf16* __restrict__ xt) {
  __shared__ float tile[32][33];
  int bidx = blockIdx.x;            // 16384 blocks
  int hwt = bidx & 31;
  int ct  = (bidx >> 5) & 31;
  int b   = bidx >> 10;
  int tid = threadIdx.x;
  int lx = tid & 31, ly = tid >> 5;
#pragma unroll
  for (int j = 0; j < 4; ++j) {
    int c = ct * 32 + ly + j * 8;
    tile[ly + j * 8][lx] = x[(size_t)(b * CIN + c) * HW + hwt * 32 + lx];
  }
  __syncthreads();
#pragma unroll
  for (int j = 0; j < 2; ++j) {
    int p = j * 256 + tid;
    int hw_l = p >> 4;
    int cp   = (p & 15) * 2;
    int m = (b << 10) + hwt * 32 + hw_l;
    *(unsigned*)&xt[(size_t)m * CIN + ct * 32 + cp] = pack2(tile[cp][hw_l], tile[cp + 1][hw_l]);
  }
}

// ============================ stage 1: conv1x1 + BN1 + ReLU ============================
__global__ __launch_bounds__(256) void k_conv1(const bf16* __restrict__ xt,
    const bf16* __restrict__ B1t, const float* __restrict__ s1,
    const float* __restrict__ t1, bf16* __restrict__ h1) {
  __shared__ __align__(16) char smem[MT * SP * 2];   // 34816B >= 4x8KB gemm bufs
  bf16* A0 = (bf16*)smem;
  bf16* A1 = (bf16*)(smem + 8192);
  bf16* B0 = (bf16*)(smem + 16384);
  bf16* B1 = (bf16*)(smem + 24576);
  const int tid = threadIdx.x;
  const int m0 = blockIdx.x * MT, n0 = blockIdx.y * 128;
  v8f acc[2][4] = {};
  gemm_db<4>(xt, B1t, m0, n0, CIN, tid, A0, A1, B0, B1, acc);
  __syncthreads();                                   // frag reads done; reuse smem
  epilogue_bnrelu<4>(acc, (bf16*)smem, s1, t1, h1, m0, n0, tid);
}

// ============================ stage 2: offset conv (3x3, 256->18) ============================
__global__ __launch_bounds__(256) void k_offset(const bf16* __restrict__ h1,
    const float* __restrict__ off_w, const float* __restrict__ off_b,
    float* __restrict__ off) {
  int t = blockIdx.x * 256 + threadIdx.x;
  if (t >= 16 * 18 * HW) return;
  int hw = t & 1023;
  int o  = (t >> 10) % 18;
  int b  = t / (18 * HW);
  int y = hw >> 5, x = hw & 31;
  float acc = off_b[o];
  for (int ky = 0; ky < 3; ++ky) {
    int yy = y + ky - 1; if (yy < 0 || yy > 31) continue;
    for (int kx = 0; kx < 3; ++kx) {
      int xx = x + kx - 1; if (xx < 0 || xx > 31) continue;
      const bf16*  hp = h1 + (size_t)((b << 10) + (yy << 5) + xx) * P_;
      const float* wp = off_w + (size_t)o * P_ * 9 + (ky * 3 + kx);
      float a = 0.f;
#pragma unroll 4
      for (int c = 0; c < P_; c += 4) {
        unsigned long long u = *(const unsigned long long*)(hp + c);
        unsigned u0 = (unsigned)u, u1 = (unsigned)(u >> 32);
        a += bf_lo(u0) * wp[(size_t)c * 9]       + bf_hi(u0) * wp[(size_t)(c + 1) * 9]
           + bf_lo(u1) * wp[(size_t)(c + 2) * 9] + bf_hi(u1) * wp[(size_t)(c + 3) * 9];
      }
      acc += a;
    }
  }
  off[t] = acc;
}

// ============================ stage 3: deformable conv GEMM + BN2 + ReLU ============================
__global__ __launch_bounds__(256) void k_deform(const bf16* __restrict__ h1,
    const float* __restrict__ off, const bf16* __restrict__ B2t,
    const float* __restrict__ s2, const float* __restrict__ t2,
    bf16* __restrict__ h2) {
  __shared__ __align__(16) char smem[61440];         // A 8K | B0 8K | B1 8K | sw 18K | si 18K
  bf16*  ldsA = (bf16*)smem;
  bf16*  B0   = (bf16*)(smem + 8192);
  bf16*  B1   = (bf16*)(smem + 16384);
  float* sw   = (float*)(smem + 24576);
  int*   si   = (int*)(smem + 43008);
  const int tid = threadIdx.x;
  const int lane = tid & 31, wv = tid >> 5;
  const int mig = wv & 3, nig = wv >> 2;
  const int m0 = blockIdx.x * MT, n0 = blockIdx.y * 128;
  const int b = m0 >> 10;

  for (int r = tid; r < MT * 9; r += 256) {          // bilinear sampling records
    int mm = r / 9, kidx = r % 9;
    int hw = (m0 + mm) & 1023;
    int y = hw >> 5, xx = hw & 31;
    float oy = off[((size_t)(b * 18 + 2 * kidx)     << 10) + hw];
    float ox = off[((size_t)(b * 18 + 2 * kidx + 1) << 10) + hw];
    float py = (float)(y + kidx / 3 - 1) + oy;
    float px = (float)(xx + kidx % 3 - 1) + ox;
    float y0f = floorf(py), x0f = floorf(px);
    float wy = py - y0f, wx = px - x0f;
    int y0 = (int)y0f, x0 = (int)x0f;
#pragma unroll
    for (int d = 0; d < 4; ++d) {
      int dy = d >> 1, dx = d & 1;
      int yi = y0 + dy, xi = x0 + dx;
      bool valid = (yi >= 0) & (yi <= 31) & (xi >= 0) & (xi <= 31);
      float wgt = (dy ? wy : 1.f - wy) * (dx ? wx : 1.f - wx);
      int yic = min(max(yi, 0), 31), xic = min(max(xi, 0), 31);
      sw[(mm * 9 + kidx) * 4 + d] = valid ? wgt : 0.f;
      si[(mm * 9 + kidx) * 4 + d] = (b << 10) + (yic << 5) + xic;
    }
  }

  v8f acc[2][4] = {};
#if USE_TDM
  if (tid < 32) tdm_load_2d(B0, B2t + (size_t)n0 * K2, 128, 32, K2);
#else
  issue_tile<128>(B0, B2t, n0, K2, 0, tid);
#endif
  const int nch = K2 / 32;                           // 72
#pragma unroll 2
  for (int i = 0; i < nch; ++i) {
    bf16* cB = (i & 1) ? B1 : B0;
    bf16* nB = (i & 1) ? B0 : B1;
    int kidx = i >> 3, c0 = (i & 7) * 32;
    __syncthreads();
    if (i + 1 < nch) {
#if USE_TDM
      if (tid < 32) tdm_load_2d(nB, B2t + (size_t)n0 * K2 + (i + 1) * 32, 128, 32, K2);
#else
      issue_tile<128>(nB, B2t, n0, K2, (i + 1) * 32, tid);
#endif
    }
    // A tile: gather+blend, 4 channels per thread per step (b64 gathers)
#pragma unroll
    for (int j = 0; j < 4; ++j) {
      int idx = j * 256 + tid;
      int mm = idx >> 3;
      int kp = (idx & 7) * 4;
      const float* w4 = &sw[(mm * 9 + kidx) * 4];
      const int*   i4 = &si[(mm * 9 + kidx) * 4];
      int c = c0 + kp;
      float f0 = 0.f, f1 = 0.f, f2 = 0.f, f3 = 0.f;
#pragma unroll
      for (int d = 0; d < 4; ++d) {
        unsigned long long u = *(const unsigned long long*)(h1 + (size_t)i4[d] * P_ + c);
        unsigned u0 = (unsigned)u, u1 = (unsigned)(u >> 32);
        float w = w4[d];
        f0 += w * bf_lo(u0); f1 += w * bf_hi(u0);
        f2 += w * bf_lo(u1); f3 += w * bf_hi(u1);
      }
      unsigned long long rr = ((unsigned long long)pack2(f2, f3) << 32) | pack2(f0, f1);
      *(unsigned long long*)&ldsA[mm * 32 + kp] = rr;
    }
#if USE_TDM
    if (tid < 32) {
      if (i + 1 < nch) __builtin_amdgcn_s_wait_tensorcnt(1);
      else             __builtin_amdgcn_s_wait_tensorcnt(0);
    }
#else
    if (i + 1 < nch) wait_async_n<2>(); else wait_async_n<0>();
#endif
    __syncthreads();
    mma_tile<4>(ldsA, cB, lane, mig, nig, acc);
  }
  __syncthreads();                                   // frag reads done; reuse smem
  epilogue_bnrelu<4>(acc, (bf16*)smem, s2, t2, h2, m0, n0, tid);
}

// ============================ stage 4: conv1x1 + BN3 + shortcut + ReLU ============================
__global__ __launch_bounds__(256) void k_conv3(const bf16* __restrict__ h2,
    const bf16* __restrict__ B3t, const float* __restrict__ s3,
    const float* __restrict__ t3, const float* __restrict__ x,
    float* __restrict__ out) {
  __shared__ __align__(16) bf16 A0[MT * 32], A1[MT * 32];
  __shared__ __align__(16) bf16 B0[64 * 32], B1[64 * 32];
  __shared__ float lout[MT * 65];        // padded for bank-conflict-free transpose
  const int tid = threadIdx.x;
  const int lane = tid & 31, wv = tid >> 5;
  const int mig = wv & 3, nig = wv >> 2;
  const int m0 = blockIdx.x * MT, n0 = blockIdx.y * 64;
  const int b = m0 >> 10, hw0 = m0 & 1023;
  v8f acc[2][2] = {};
  gemm_db<2>(h2, B3t, m0, n0, P_, tid, A0, A1, B0, B1, acc);
  int hl = lane >> 4, col = lane & 15;
#pragma unroll
  for (int q = 0; q < 2; ++q) {
    int cl = (nig * 2 + q) * 16 + col;
    float s = s3[n0 + cl], t = t3[n0 + cl];
#pragma unroll
    for (int p = 0; p < 2; ++p)
#pragma unroll
      for (int r = 0; r < 8; ++r) {
        int rl = (mig * 2 + p) * 16 + r + 8 * hl;
        lout[rl * 65 + cl] = acc[p][q][r] * s + t;
      }
  }
  __syncthreads();
#pragma unroll
  for (int j = 0; j < 32; ++j) {                 // hw-contiguous writeback + shortcut
    int idx = j * 256 + tid;
    int hw_l = idx & 127, n_l = idx >> 7;
    size_t gi = (size_t)(b * O_ + n0 + n_l) * HW + hw0 + hw_l;
    out[gi] = fmaxf(lout[hw_l * 65 + n_l] + x[gi], 0.f);
  }
}

// ============================ launch ============================
extern "C" void kernel_launch(void* const* d_in, const int* in_sizes, int n_in,
                              void* d_out, int out_size, void* d_ws, size_t ws_size,
                              hipStream_t stream) {
  const float* x     = (const float*)d_in[0];
  const float* w1    = (const float*)d_in[1];
  const float* bn1g  = (const float*)d_in[2];
  const float* bn1b  = (const float*)d_in[3];
  const float* bn1m  = (const float*)d_in[4];
  const float* bn1v  = (const float*)d_in[5];
  const float* off_w = (const float*)d_in[6];
  const float* off_b = (const float*)d_in[7];
  const float* w2    = (const float*)d_in[8];
  const float* bn2g  = (const float*)d_in[9];
  const float* bn2b  = (const float*)d_in[10];
  const float* bn2m  = (const float*)d_in[11];
  const float* bn2v  = (const float*)d_in[12];
  const float* w3    = (const float*)d_in[13];
  const float* bn3g  = (const float*)d_in[14];
  const float* bn3b  = (const float*)d_in[15];
  const float* bn3m  = (const float*)d_in[16];
  const float* bn3v  = (const float*)d_in[17];

  char* ws = (char*)d_ws;
  bf16*  xt  = (bf16*)(ws + 0);                  // 33.5 MB
  bf16*  h1  = (bf16*)(ws + 33554432);           //  8 MB
  bf16*  h2  = (bf16*)(ws + 41943040);           //  8 MB
  float* off = (float*)(ws + 50331648);          //  1.125 MB
  bf16*  B1t = (bf16*)(ws + 51511296);           //  0.5 MB
  bf16*  B2t = (bf16*)(ws + 52035584);           //  1.125 MB
  bf16*  B3t = (bf16*)(ws + 53215232);           //  0.5 MB
  float* s1  = (float*)(ws + 53739520);
  float* t1 = s1 + 256; float* s2 = t1 + 256; float* t2 = s2 + 256;
  float* s3 = t2 + 256; float* t3 = s3 + 1024;

  k_bnprep<<<6, 256, 0, stream>>>(bn1g, bn1b, bn1m, bn1v,
                                  bn2g, bn2b, bn2m, bn2v,
                                  bn3g, bn3b, bn3m, bn3v,
                                  s1, t1, s2, t2, s3, t3);
  k_castw <<<1024, 256, 0, stream>>>(w1, B1t, 256 * 1024);
  k_packw2<<<2304, 256, 0, stream>>>(w2, B2t);
  k_castw <<<1024, 256, 0, stream>>>(w3, B3t, 1024 * 256);
  k_xt    <<<16384, 256, 0, stream>>>(x, xt);

  k_conv1 <<<dim3(128, 2),  256, 0, stream>>>(xt, B1t, s1, t1, h1);
  k_offset<<<1152,          256, 0, stream>>>(h1, off_w, off_b, off);
  k_deform<<<dim3(128, 2),  256, 0, stream>>>(h1, off, B2t, s2, t2, h2);
  k_conv3 <<<dim3(128, 16), 256, 0, stream>>>(h2, B3t, s3, t3, x, (float*)d_out);
}
